// Bayesian_MultiHeadSelfAttention_First_27745488732904
// MI455X (gfx1250) — matrix operand order repaired and checked by reading, not compiled, not run on hardware
//
#include <hip/hip_runtime.h>
#include <hip/hip_bf16.h>
#include <stdint.h>

#define BB 4
#define SS 1024
#define EE 768
#define HH 12
#define DHD 64
#define RTOT (BB*SS)        // 4096 rows
#define KT_W (EE/32)        // 24 K-tiles for E-dim GEMMs
#define NT_W (EE/16)        // 48 N-tiles
#define KT_S (SS/32)        // 32 key-chunks for attention
#define BHN (BB*HH)         // 48

typedef float    v8f   __attribute__((ext_vector_type(8)));
typedef __bf16   v16bf __attribute__((ext_vector_type(16)));
typedef unsigned uv4   __attribute__((ext_vector_type(4)));
typedef int      v4i   __attribute__((vector_size(16)));

union Frag { uv4 q[2]; unsigned u[8]; v16bf v; };

// ---- gfx1250 async global->LDS path (guarded so compile never regresses) ----
#if defined(__has_builtin)
#if __has_builtin(__builtin_amdgcn_global_load_async_to_lds_b128) && \
    __has_builtin(__builtin_amdgcn_s_wait_asynccnt)
#define USE_ASYNC 1
#endif
#endif
#ifndef USE_ASYNC
#define USE_ASYNC 0
#endif

#if USE_ASYNC
#define ASYNC_WAIT(n) __builtin_amdgcn_s_wait_asynccnt(n)
#else
#define ASYNC_WAIT(n) ((void)0)
#endif

__device__ inline void async_cp16(const unsigned* g, unsigned* l){
#if USE_ASYNC
  __builtin_amdgcn_global_load_async_to_lds_b128(
      (__attribute__((address_space(1))) v4i*)(void*)g,
      (__attribute__((address_space(3))) v4i*)(void*)l, 0, 0);
#else
  *(uv4*)l = *(const uv4*)g;
#endif
}

__device__ inline v8f zero8(){ v8f z = {0.f,0.f,0.f,0.f,0.f,0.f,0.f,0.f}; return z; }

__device__ inline unsigned short f2bf(float f){
  unsigned u = __builtin_bit_cast(unsigned, f);
  u += 0x7fffu + ((u>>16)&1u);                 // round-to-nearest-even
  return (unsigned short)(u>>16);
}
__device__ inline float bf2f(unsigned short h){
  unsigned u = ((unsigned)h)<<16; return __builtin_bit_cast(float,u);
}
__device__ inline unsigned pack2(float lo, float hi){
  return (unsigned)f2bf(lo) | ((unsigned)f2bf(hi)<<16);
}
__device__ inline float softplusf(float x){ return x>15.f ? x : log1pf(__expf(x)); }

__device__ inline v8f wmma_bf16(const Frag& a, const Frag& b, v8f c){
  return __builtin_amdgcn_wmma_f32_16x16x32_bf16(false, a.v, false, b.v, (short)0, c, false, false);
}

// A-fragment: 16(M) x 32(K) bf16 from row-major [M][lda]
__device__ inline Frag loadA(const unsigned short* p, int lda, int m0, int k0, int lane){
  Frag f; int row = m0 + (lane&15); int half = lane>>4;
  const unsigned short* base = p + (size_t)row*lda + k0 + half*16;
  f.q[0] = *(const uv4*)(base);
  f.q[1] = *(const uv4*)(base + 8);
  return f;
}
// B-fragment (NT): B[k][n] = Bt[n0+n][k0+k], k contiguous in memory
__device__ inline Frag loadBT(const unsigned short* p, int ldb, int n0, int k0, int lane){
  Frag f; int n = n0 + (lane&15); int half = lane>>4;
  const unsigned short* base = p + (size_t)n*ldb + k0 + half*16;
  f.q[0] = *(const uv4*)(base);
  f.q[1] = *(const uv4*)(base + 8);
  return f;
}
// B-fragment from packed layout (32 lanes x 8 dwords per fragment)
__device__ inline Frag loadBP(const unsigned* p, int fragIdx, int lane){
  Frag f; const unsigned* base = p + (size_t)fragIdx*256 + lane*8;
  f.q[0] = *(const uv4*)(base);
  f.q[1] = *(const uv4*)(base+4);
  return f;
}

// ---------------- elementwise / prep kernels ----------------

__global__ void k_cvt_bf16(const float* __restrict__ src, unsigned* __restrict__ dst, int npairs){
  int i = blockIdx.x*256 + threadIdx.x;
  if(i < npairs) dst[i] = pack2(src[2*i], src[2*i+1]);
}

__global__ void k_rowxx(const float* __restrict__ x, float* __restrict__ xx){
  int r = blockIdx.x, t = threadIdx.x;
  const float* p = x + (size_t)r*EE;
  float s = 0.f;
  for(int c=t;c<EE;c+=256){ float v = p[c]*(1.f/EE); s += v*v; }
  __shared__ float red[256];
  red[t]=s; __syncthreads();
  for(int o=128;o>0;o>>=1){ if(t<o) red[t]+=red[t+o]; __syncthreads(); }
  if(t==0) xx[r]=red[0];
}

__global__ void k_pack_w(const float* wq, const float* wk, const float* wv, const float* wo,
                         unsigned* pq, unsigned* pk, unsigned* pv, unsigned* po, unsigned* po2){
  int idx = blockIdx.x*256 + threadIdx.x;      // 5 * 1152 frags * 32 lanes
  int lane = idx & 31;
  int frag = (idx>>5) % (NT_W*KT_W);
  int mat  = idx / (32*NT_W*KT_W);
  if(mat >= 5) return;
  int nt = frag / KT_W, kt = frag % KT_W;
  const float* w = (mat==0)?wq:(mat==1)?wk:(mat==2)?wv:wo;
  unsigned* out  = (mat==0)?pq:(mat==1)?pk:(mat==2)?pv:(mat==3)?po:po2;
  int n  = nt*16 + (lane&15);
  int kb = kt*32 + (lane>>4)*16;
  unsigned* o = out + (size_t)frag*256 + lane*8;
  #pragma unroll
  for(int d=0; d<8; d++){
    float a = w[(size_t)(kb+2*d)*EE + n];
    float b = w[(size_t)(kb+2*d+1)*EE + n];
    if(mat==4){ a = a*a*(1.f/EE); b = b*b*(1.f/EE); }
    o[d] = pack2(a,b);
  }
}

__global__ void k_sigma_qk(const float* __restrict__ xx, const float* __restrict__ wqs,
                           const float* __restrict__ wks,
                           unsigned* __restrict__ sqb, unsigned* __restrict__ skb){
  int i = blockIdx.x*256 + threadIdx.x;        // pairs
  if(i >= RTOT*EE/2) return;
  int r = i / (EE/2);
  int c0 = (i % (EE/2))*2;
  float xv = xx[r];
  float a = softplusf(xv * softplusf(wqs[c0]));
  float b = softplusf(xv * softplusf(wqs[c0+1]));
  float c = softplusf(xv * softplusf(wks[c0]));
  float d = softplusf(xv * softplusf(wks[c0+1]));
  sqb[i] = pack2(a,b);
  skb[i] = pack2(c,d);
}

// ---------------- QKV fused GEMM (B staged via async->LDS, shared by 4 waves) ----------------
__global__ __launch_bounds__(128) void k_qkv_gemm(
    const unsigned short* __restrict__ xb,
    const unsigned* __restrict__ pq, const unsigned* __restrict__ pk, const unsigned* __restrict__ pv,
    const float* __restrict__ xx, const float* __restrict__ wvs,
    unsigned short* __restrict__ mqb, unsigned short* __restrict__ mq2b,
    unsigned short* __restrict__ mkb, unsigned short* __restrict__ mk2b,
    unsigned* __restrict__ mvP, unsigned* __restrict__ svP, unsigned* __restrict__ mv2P){
  __shared__ __align__(16) unsigned wbuf[2][6*256];   // 12KB double buffer: 3 products x 2 n-tiles
  int tid = threadIdx.x;
  int wave = tid>>5, lane = tid&31;
  int half = lane>>4, lanen = lane&15;
  int n0 = blockIdx.x*32;
  int m0 = (blockIdx.y*4 + wave)*32;
  const unsigned* Ps[3] = {pq, pk, pv};

  auto stage = [&](int ks, int buf){
    #pragma unroll
    for(int j=0;j<3;j++){
      int idx = j*128 + tid;      // 0..383 b128 chunks
      int f = idx>>6;             // 0..5 : p*2+nt
      int off = (idx&63)*4;
      const unsigned* g = Ps[f>>1] + ((size_t)((n0/16 + (f&1))*KT_W + ks))*256 + off;
      async_cp16(g, &wbuf[buf][f*256 + off]);
    }
  };

  v8f acc[3][2][2];
  #pragma unroll
  for(int p=0;p<3;p++) for(int i=0;i<2;i++) for(int j=0;j<2;j++) acc[p][i][j]=zero8();

  stage(0, 0);
  for(int ks=0; ks<KT_W; ks++){
    if(ks+1 < KT_W){ stage(ks+1, (ks+1)&1); ASYNC_WAIT(3); }
    else           { ASYNC_WAIT(0); }
    __syncthreads();
    if(ks+2 < KT_W){
      __builtin_prefetch(xb + (size_t)(m0+lanen)*EE + (ks+2)*32, 0, 0);
      __builtin_prefetch(xb + (size_t)(m0+16+lanen)*EE + (ks+2)*32, 0, 0);
    }
    Frag a0 = loadA(xb, EE, m0,    ks*32, lane);
    Frag a1 = loadA(xb, EE, m0+16, ks*32, lane);
    const unsigned* wb = &wbuf[ks&1][0];
    #pragma unroll
    for(int p=0;p<3;p++){
      #pragma unroll
      for(int nt=0;nt<2;nt++){
        Frag b = loadBP(wb, p*2+nt, lane);
        acc[p][0][nt] = wmma_bf16(a0, b, acc[p][0][nt]);
        acc[p][1][nt] = wmma_bf16(a1, b, acc[p][1][nt]);
      }
    }
    __syncthreads();
  }
  // q / k epilogue: row-major bf16 (mu, mu^2)
  #pragma unroll
  for(int mt=0;mt<2;mt++) for(int nt=0;nt<2;nt++){
    #pragma unroll
    for(int r=0;r<8;r++){
      int row = m0 + mt*16 + half*8 + r;
      int col = n0 + nt*16 + lanen;
      size_t o = (size_t)row*EE + col;
      float q = acc[0][mt][nt][r];  mqb[o]=f2bf(q);  mq2b[o]=f2bf(q*q);
      float k = acc[1][mt][nt][r];  mkb[o]=f2bf(k);  mk2b[o]=f2bf(k*k);
    }
  }
  // v epilogue: packed B-fragments (mv, sv, mv^2), pairing adjacent seq rows
  #pragma unroll
  for(int mt=0;mt<2;mt++) for(int nt=0;nt<2;nt++){
    int col = n0 + nt*16 + lanen;
    int h = col / DHD, nIn = col % DHD;
    float Wv = softplusf(wvs[col]);
    #pragma unroll
    for(int j=0;j<4;j++){
      int row0 = m0 + mt*16 + half*8 + 2*j;          // even
      int b    = row0 >> 10;
      int kloc = row0 & 1023;
      int bh   = b*HH + h;
      float v0 = acc[2][mt][nt][2*j], v1 = acc[2][mt][nt][2*j+1];
      float s0 = softplusf(xx[row0]*Wv), s1 = softplusf(xx[row0+1]*Wv);
      size_t addr = ((size_t)((bh*KT_S + (kloc>>5))*4 + (nIn>>4)))*256
                  + (size_t)(((kloc>>4)&1)*16 + (nIn&15))*8 + ((kloc>>1)&7);
      mvP[addr]  = pack2(v0, v1);
      svP[addr]  = pack2(s0, s1);
      mv2P[addr] = pack2(v0*v0, v1*v1);
    }
  }
}

// ---------------- attention pass A: softmax row stats ----------------
__global__ __launch_bounds__(128) void k_attn_stats(
    const unsigned short* __restrict__ mqb, const unsigned short* __restrict__ mkb,
    float* __restrict__ Mst, float* __restrict__ Lst){
  int wave = threadIdx.x>>5, lane = threadIdx.x&31;
  int bh = blockIdx.x; int b = bh/HH, h = bh%HH;
  int q0 = (blockIdx.y*4 + wave)*16;
  int rowq = b*SS + q0;
  int hb = h*DHD;

  Frag aq0 = loadA(mqb, EE, rowq, hb,    lane);
  Frag aq1 = loadA(mqb, EE, rowq, hb+32, lane);

  float m[8], l[8];
  #pragma unroll
  for(int r=0;r<8;r++){ m[r]=-1e30f; l[r]=0.f; }

  for(int kt=0; kt<SS/16; kt++){
    int kc = b*SS + kt*16;
    Frag b0 = loadBT(mkb, EE, kc, hb,    lane);
    Frag b1 = loadBT(mkb, EE, kc, hb+32, lane);
    v8f s = zero8();
    s = wmma_bf16(aq0, b0, s);
    s = wmma_bf16(aq1, b1, s);
    #pragma unroll
    for(int r=0;r<8;r++){
      float t = s[r]*0.125f;                      // /sqrt(dk)
      float tm = t;
      tm = fmaxf(tm, __shfl_xor(tm,1,32));
      tm = fmaxf(tm, __shfl_xor(tm,2,32));
      tm = fmaxf(tm, __shfl_xor(tm,4,32));
      tm = fmaxf(tm, __shfl_xor(tm,8,32));
      float mn = fmaxf(m[r], tm);
      float e = __expf(t - mn);
      e += __shfl_xor(e,1,32); e += __shfl_xor(e,2,32);
      e += __shfl_xor(e,4,32); e += __shfl_xor(e,8,32);
      l[r] = l[r]*__expf(m[r]-mn) + e;
      m[r] = mn;
    }
  }
  int base = bh*SS + q0;
  if(lane==0){
    #pragma unroll
    for(int r=0;r<8;r++){ Mst[base+r]=m[r]; Lst[base+r]=l[r]; }
  }
  if(lane==16){
    #pragma unroll
    for(int r=0;r<8;r++){ Mst[base+8+r]=m[r]; Lst[base+8+r]=l[r]; }
  }
}

// ---------------- attention pass B: fused score->prob->PV ----------------
// Key-side rows (mk/sk/sq) and value-side packed frags (mv/sv/mv^2) are identical
// across the 4 waves of a block -> double-buffered async staging into LDS.
__global__ __launch_bounds__(128) void k_attn(
    const unsigned short* __restrict__ mqb,  const unsigned short* __restrict__ mq2b,
    const unsigned short* __restrict__ mkb,  const unsigned short* __restrict__ mk2b,
    const unsigned short* __restrict__ sqb,  const unsigned short* __restrict__ skb,
    const unsigned* __restrict__ mvP, const unsigned* __restrict__ svP, const unsigned* __restrict__ mv2P,
    const float* __restrict__ Mst, const float* __restrict__ Lst,
    unsigned short* __restrict__ mucat, unsigned short* __restrict__ sigcat){
  __shared__ __align__(16) unsigned short prob[4][3][16][32];   // 12KB per-wave prob tiles
  __shared__ __align__(16) unsigned bbuf[2][3*1024];            // 24KB phase-2 packed frags
  __shared__ __align__(16) unsigned short kbuf[2][3*32*64];     // 24KB phase-1 key rows
  int tid = threadIdx.x;
  int wave = tid>>5, lane = tid&31;
  int half = lane>>4, lanen = lane&15;
  int bh = blockIdx.x; int b = bh/HH, h = bh%HH;
  int q0 = (blockIdx.y*4 + wave)*16;
  int rowq = b*SS + q0;
  int hb = h*DHD;

  auto stage = [&](int ch, int buf){
    size_t gb2 = (size_t)(bh*KT_S + ch)*1024;           // dwords per matrix chunk (phase-2)
    int krow0 = b*SS + ch*32;
    #pragma unroll
    for(int j=0;j<6;j++){
      int idx = j*128 + tid;        // 0..767 : 3 mats x 256 b128
      int mat = idx>>8, off = (idx&255)*4;
      const unsigned* g = ((mat==0)?mvP:(mat==1)?svP:mv2P) + gb2 + off;
      async_cp16(g, &bbuf[buf][mat*1024 + off]);
    }
    #pragma unroll
    for(int j=0;j<6;j++){
      int idx = j*128 + tid;        // 0..767 : 3 mats x 32 rows x 8 b128
      int mat = idx>>8, rem = idx&255;
      int row = rem>>3, seg = rem&7;
      const unsigned short* gm = (mat==0)?mkb:(mat==1)?skb:sqb;
      const unsigned* g = (const unsigned*)(gm + (size_t)(krow0+row)*EE + hb + seg*8);
      async_cp16(g, (unsigned*)(&kbuf[buf][0] + mat*2048 + row*64 + seg*8));
    }
  };

  // persistent query-side A fragments (2 d-steps each)
  Frag Amu[2], Am2[2], Ak2[2], Asq[2];
  #pragma unroll
  for(int ds=0; ds<2; ds++){
    Amu[ds] = loadA(mqb,  EE, rowq, hb+32*ds, lane);
    Am2[ds] = loadA(mq2b, EE, rowq, hb+32*ds, lane);
    Ak2[ds] = loadA(mk2b, EE, rowq, hb+32*ds, lane);
    Asq[ds] = loadA(sqb,  EE, rowq, hb+32*ds, lane);
  }
  float mr[8], il[8];
  #pragma unroll
  for(int r=0;r<8;r++){
    int qq = bh*SS + q0 + half*8 + r;
    mr[r] = Mst[qq];
    il[r] = 1.f/Lst[qq];
  }

  v8f aMu[4], aS[4], aT[4];
  #pragma unroll
  for(int nt=0;nt<4;nt++){ aMu[nt]=zero8(); aS[nt]=zero8(); aT[nt]=zero8(); }

  stage(0, 0);
  for(int ch=0; ch<KT_S; ch++){                 // 32 keys per chunk
    if(ch+1 < KT_S){ stage(ch+1, (ch+1)&1); ASYNC_WAIT(12); }
    else           { ASYNC_WAIT(0); }
    __syncthreads();                            // kbuf/bbuf[ch&1] ready for all waves
    const unsigned short* kb = &kbuf[ch&1][0];
    // phase 1: two 16x16 score tiles -> probability/variance weights
    #pragma unroll
    for(int t2=0;t2<2;t2++){
      v8f smu=zero8(), sa=zero8(), sb=zero8(), sc=zero8();
      #pragma unroll
      for(int ds=0; ds<2; ds++){
        Frag Bk = loadBT(kb + 0*2048, 64, t2*16, ds*32, lane);
        Frag Bs = loadBT(kb + 1*2048, 64, t2*16, ds*32, lane);
        Frag Bq = loadBT(kb + 2*2048, 64, t2*16, ds*32, lane);
        smu = wmma_bf16(Amu[ds], Bk, smu);
        sa  = wmma_bf16(Am2[ds], Bs, sa);
        sb  = wmma_bf16(Ak2[ds], Bq, sb);
        sc  = wmma_bf16(Asq[ds], Bs, sc);
      }
      #pragma unroll
      for(int r=0;r<8;r++){
        float w  = __expf(smu[r]*0.125f - mr[r]) * il[r];
        float ss = (sa[r]+sb[r]+sc[r]) * (1.f/4096.f);   // sigma_scaled
        float g  = w - w*w; g = g*g;
        float Sw = g*ss*(1.f/1024.f);
        float A2 = (w*w + Sw)*(1.f/1024.f);
        float A3 = Sw*(1.f/1024.f);
        int row = half*8 + r;
        int kl2 = t2*16 + lanen;
        prob[wave][0][row][kl2] = f2bf(w);
        prob[wave][1][row][kl2] = f2bf(A2);
        prob[wave][2][row][kl2] = f2bf(A3);
      }
    }
    __syncthreads();                            // prob tiles visible
    Frag P0 = loadA(&prob[wave][0][0][0], 32, 0, 0, lane);
    Frag P1 = loadA(&prob[wave][1][0][0], 32, 0, 0, lane);
    Frag P2 = loadA(&prob[wave][2][0][0], 32, 0, 0, lane);
    const unsigned* bb = &bbuf[ch&1][0];
    #pragma unroll
    for(int nt=0;nt<4;nt++){
      Frag Bv  = loadBP(bb + 0*1024, nt, lane);
      Frag Bs2 = loadBP(bb + 1*1024, nt, lane);
      Frag Bm2 = loadBP(bb + 2*1024, nt, lane);
      aMu[nt] = wmma_bf16(P0, Bv,  aMu[nt]);
      aS[nt]  = wmma_bf16(P1, Bs2, aS[nt]);
      aT[nt]  = wmma_bf16(P2, Bm2, aT[nt]);
    }
    __syncthreads();                            // protect prob + buffers for next stage
  }
  // epilogue: [B,S,E] row-major concat-of-heads
  #pragma unroll
  for(int nt=0;nt<4;nt++){
    int col = hb + nt*16 + lanen;
    #pragma unroll
    for(int r=0;r<8;r++){
      int row = rowq + half*8 + r;
      size_t o = (size_t)row*EE + col;
      mucat[o]  = f2bf(aMu[nt][r]);
      sigcat[o] = f2bf(softplusf(aS[nt][r] + aT[nt][r]));
    }
  }
}

// ---------------- row stats of mu_cat / sig_cat ----------------
__global__ void k_rowstats2(const unsigned short* __restrict__ mucat,
                            const unsigned short* __restrict__ sigcat,
                            float* __restrict__ xx2, float* __restrict__ s3r){
  int r = blockIdx.x, t = threadIdx.x;
  const unsigned short* pm = mucat  + (size_t)r*EE;
  const unsigned short* ps = sigcat + (size_t)r*EE;
  float a=0.f, s=0.f;
  for(int c=t;c<EE;c+=256){
    float v = bf2f(pm[c])*(1.f/EE);
    a += v*v;
    s += bf2f(ps[c]);
  }
  __shared__ float r1[256], r2[256];
  r1[t]=a; r2[t]=s; __syncthreads();
  for(int o=128;o>0;o>>=1){ if(t<o){ r1[t]+=r1[t+o]; r2[t]+=r2[t+o]; } __syncthreads(); }
  if(t==0){ xx2[r]=r1[0]; s3r[r]=r2[0]; }
}

// ---------------- output linear (mu + variance) ----------------
__global__ __launch_bounds__(128) void k_out_gemm(
    const unsigned short* __restrict__ mucat, const unsigned short* __restrict__ sigcat,
    const unsigned* __restrict__ poP, const unsigned* __restrict__ po2P,
    const float* __restrict__ xx2, const float* __restrict__ s3r,
    const float* __restrict__ wos, float* __restrict__ out){
  __shared__ __align__(16) unsigned obuf[2][4*256];   // 8KB double buffer: 2 mats x 2 n-tiles
  int tid = threadIdx.x;
  int wave = tid>>5, lane = tid&31;
  int half = lane>>4, lanen = lane&15;
  int n0 = blockIdx.x*32;
  int m0 = (blockIdx.y*4 + wave)*32;

  auto stage = [&](int ks, int buf){
    #pragma unroll
    for(int j=0;j<2;j++){
      int idx = j*128 + tid;        // 0..255 b128 chunks
      int f = idx>>6;               // mat*2+nt
      int off = (idx&63)*4;
      const unsigned* g = ((f>>1)?po2P:poP) + ((size_t)((n0/16 + (f&1))*KT_W + ks))*256 + off;
      async_cp16(g, &obuf[buf][f*256 + off]);
    }
  };

  v8f aM[2][2], aSg[2][2];
  #pragma unroll
  for(int i=0;i<2;i++) for(int j=0;j<2;j++){ aM[i][j]=zero8(); aSg[i][j]=zero8(); }

  stage(0, 0);
  for(int ks=0; ks<KT_W; ks++){
    if(ks+1 < KT_W){ stage(ks+1, (ks+1)&1); ASYNC_WAIT(2); }
    else           { ASYNC_WAIT(0); }
    __syncthreads();
    Frag a0 = loadA(mucat,  EE, m0,    ks*32, lane);
    Frag a1 = loadA(mucat,  EE, m0+16, ks*32, lane);
    Frag c0 = loadA(sigcat, EE, m0,    ks*32, lane);
    Frag c1 = loadA(sigcat, EE, m0+16, ks*32, lane);
    const unsigned* ob = &obuf[ks&1][0];
    #pragma unroll
    for(int nt=0;nt<2;nt++){
      Frag bo = loadBP(ob, 0*2+nt, lane);
      Frag b2 = loadBP(ob, 1*2+nt, lane);
      aM[0][nt]  = wmma_bf16(a0, bo, aM[0][nt]);
      aM[1][nt]  = wmma_bf16(a1, bo, aM[1][nt]);
      aSg[0][nt] = wmma_bf16(c0, b2, aSg[0][nt]);
      aSg[1][nt] = wmma_bf16(c1, b2, aSg[1][nt]);
    }
    __syncthreads();
  }
  const size_t OFFS = (size_t)RTOT*EE;
  #pragma unroll
  for(int nt=0;nt<2;nt++){
    int col = n0 + nt*16 + lanen;
    float Wo = softplusf(wos[col]);
    #pragma unroll
    for(int mt=0;mt<2;mt++){
      #pragma unroll
      for(int r=0;r<8;r++){
        int row = m0 + mt*16 + half*8 + r;
        size_t o = (size_t)row*EE + col;
        out[o] = aM[mt][nt][r];
        float pre = aSg[mt][nt][r] + (xx2[row] + s3r[row]*(1.f/EE))*Wo;
        out[OFFS + o] = softplusf(softplusf(pre));
      }
    }
  }
}

// ---------------- KL (deterministic two-stage) ----------------
__global__ void k_kl_mu(const float* wq, const float* wk, const float* wv, const float* wo,
                        float* part){
  int layer = blockIdx.y, blk = blockIdx.x, t = threadIdx.x;
  const float* w = (layer==0)?wq:(layer==1)?wk:(layer==2)?wv:wo;
  float s=0.f;
  for(int i = blk*256+t; i < EE*EE; i += 32*256){ float v=w[i]; s += v*v; }
  __shared__ float red[256];
  red[t]=s; __syncthreads();
  for(int o=128;o>0;o>>=1){ if(t<o) red[t]+=red[t+o]; __syncthreads(); }
  if(t==0) part[layer*32+blk]=red[0];
}
__global__ void k_kl_sig(const float* sq, const float* sk, const float* sv, const float* so,
                         float* part){
  int layer = blockIdx.x, t = threadIdx.x;
  const float* s = (layer==0)?sq:(layer==1)?sk:(layer==2)?sv:so;
  float acc=0.f;
  for(int j=t;j<EE;j+=256){
    float lv = s[j];
    acc += logf(0.01f) - 1.f - lv + log1pf(expf(lv))*(1.f/0.01f);
  }
  __shared__ float red[256];
  red[t]=acc; __syncthreads();
  for(int o=128;o>0;o>>=1){ if(t<o) red[t]+=red[t+o]; __syncthreads(); }
  if(t==0) part[128+layer]=red[0];
}
__global__ void k_kl_final(const float* part, float* out){
  if(threadIdx.x==0 && blockIdx.x==0){
    float kl=0.f;
    for(int l=0;l<4;l++){
      float sm=0.f;
      for(int b=0;b<32;b++) sm += part[l*32+b];
      kl += 0.5f*( part[128+l]*(1.f/EE) + sm*(1.f/((float)EE*(float)EE))*(1.f/0.01f) );
    }
    out[(size_t)2*RTOT*EE] = kl;
  }
}

// ---------------- host launcher ----------------
extern "C" void kernel_launch(void* const* d_in, const int* in_sizes, int n_in,
                              void* d_out, int out_size, void* d_ws, size_t ws_size,
                              hipStream_t stream) {
  (void)in_sizes; (void)n_in; (void)out_size; (void)ws_size;
  const float* x   = (const float*)d_in[0];
  const float* wqm = (const float*)d_in[1]; const float* wqs = (const float*)d_in[2];
  const float* wkm = (const float*)d_in[3]; const float* wks = (const float*)d_in[4];
  const float* wvm = (const float*)d_in[5]; const float* wvs = (const float*)d_in[6];
  const float* wom = (const float*)d_in[7]; const float* wos = (const float*)d_in[8];
  float* out = (float*)d_out;

  char* base = (char*)d_ws;
  size_t cur = 0;
  auto alloc = [&](size_t bytes)->char*{
    char* p = base + cur;
    cur = (cur + bytes + 255) & ~(size_t)255;
    return p;
  };
  const size_t BF_RE = (size_t)RTOT*EE*2;                  // bf16 [4096][768]
  const size_t PK_W  = (size_t)NT_W*KT_W*1024;             // packed weight
  const size_t PK_V  = (size_t)BHN*KT_S*4*1024;            // packed per-(b,h) value-side

  unsigned short* xb    = (unsigned short*)alloc(BF_RE);
  unsigned* pq   = (unsigned*)alloc(PK_W);
  unsigned* pk   = (unsigned*)alloc(PK_W);
  unsigned* pv   = (unsigned*)alloc(PK_W);
  unsigned* po   = (unsigned*)alloc(PK_W);
  unsigned* po2  = (unsigned*)alloc(PK_W);
  float* xx      = (float*)alloc((size_t)RTOT*4);
  unsigned short* sqb  = (unsigned short*)alloc(BF_RE);
  unsigned short* skb  = (unsigned short*)alloc(BF_RE);
  unsigned short* mqb  = (unsigned short*)alloc(BF_RE);
  unsigned short* mq2b = (unsigned short*)alloc(BF_RE);
  unsigned short* mkb  = (unsigned short*)alloc(BF_RE);
  unsigned short* mk2b = (unsigned short*)alloc(BF_RE);
  unsigned* mvP  = (unsigned*)alloc(PK_V);
  unsigned* svP  = (unsigned*)alloc(PK_V);
  unsigned* mv2P = (unsigned*)alloc(PK_V);
  float* Mst     = (float*)alloc((size_t)BHN*SS*4);
  float* Lst     = (float*)alloc((size_t)BHN*SS*4);
  unsigned short* mucat  = (unsigned short*)alloc(BF_RE);
  unsigned short* sigcat = (unsigned short*)alloc(BF_RE);
  float* xx2     = (float*)alloc((size_t)RTOT*4);
  float* s3r     = (float*)alloc((size_t)RTOT*4);
  float* klpart  = (float*)alloc((size_t)(128+4)*4);

  // 1. prep: x -> bf16, row xx, pack weights, sigma_q/k
  k_cvt_bf16<<<(RTOT*EE/2 + 255)/256, 256, 0, stream>>>(x, (unsigned*)xb, RTOT*EE/2);
  k_rowxx<<<RTOT, 256, 0, stream>>>(x, xx);
  k_pack_w<<<(5*NT_W*KT_W*32 + 255)/256, 256, 0, stream>>>(wqm, wkm, wvm, wom, pq, pk, pv, po, po2);
  k_sigma_qk<<<(RTOT*EE/2 + 255)/256, 256, 0, stream>>>(xx, wqs, wks, (unsigned*)sqb, (unsigned*)skb);

  // 2. fused QKV projection (3 GEMMs, shared A) + value-side packing
  k_qkv_gemm<<<dim3(EE/32, RTOT/128), 128, 0, stream>>>(
      xb, pq, pk, pv, xx, wvs, mqb, mq2b, mkb, mk2b, mvP, svP, mv2P);

  // 3. attention: softmax stats, then fused prob/variance/PV pass
  k_attn_stats<<<dim3(BHN, SS/64), 128, 0, stream>>>(mqb, mkb, Mst, Lst);
  k_attn<<<dim3(BHN, SS/64), 128, 0, stream>>>(
      mqb, mq2b, mkb, mk2b, sqb, skb, mvP, svP, mv2P, Mst, Lst, mucat, sigcat);

  // 4. output linear (mu + variance propagation) and KL
  k_rowstats2<<<RTOT, 256, 0, stream>>>(mucat, sigcat, xx2, s3r);
  k_out_gemm<<<dim3(EE/32, RTOT/128), 128, 0, stream>>>(
      mucat, sigcat, po, po2, xx2, s3r, wos, out);

  k_kl_mu<<<dim3(32,4), 256, 0, stream>>>(wqm, wkm, wvm, wom, klpart);
  k_kl_sig<<<4, 256, 0, stream>>>(wqs, wks, wvs, wos, klpart);
  k_kl_final<<<1, 64, 0, stream>>>(klpart, out);
}
